// ConvSeq2Seq_64768106824392
// MI455X (gfx1250) — matrix operand order repaired and checked by reading, not compiled, not run on hardware
//
#include <hip/hip_runtime.h>
#include <hip/hip_bf16.h>

// ---------------- problem constants ----------------
constexpr int BB  = 64;        // batch
constexpr int TT  = 512;       // timesteps
constexpr int HH  = 512;       // hidden
constexpr int CSS = 64;        // c_sum (latent channels)
constexpr int KK  = 21;        // conv kernel == stride
constexpr int SEG = KK * KK;   // 441
constexpr int LL  = SEG * TT;  // 225792
constexpr int G4  = 4 * HH;    // 2048 gate columns
constexpr int NGROUPS = 16;    // persistent workgroups (each owns 32 hidden cols)
constexpr int NT2048  = G4 / 16; // 128 N-tiles for the 2048-wide weights
constexpr size_t BH = (size_t)BB * HH;

typedef __attribute__((ext_vector_type(16))) _Float16 v16h;
typedef __attribute__((ext_vector_type(8)))  _Float16 v8h;
typedef __attribute__((ext_vector_type(8)))  float    v8f;

// Opaque zero offset: defeats LICM (addresses become loop-variant so the
// t-invariant weight tiles can't be hoisted out of the 512-step recurrence
// and spilled), WITHOUT breaking the global address-space inference chain
// (pointer stays a GEP off the kernel argument -> global_load, not flat_load).
__device__ __forceinline__ size_t opaque_zero() {
  size_t z = 0;
  asm volatile("" : "+s"(z));
  return z;
}

// ---------------- WMMA fragment loaders ----------------
// A-matrix (16x32 f16) dense layout (ISA 7.12.2): lane L -> M=L%16, half=L/16;
// element e: K = (e<8 ? e : e+8) + 8*half -> two contiguous 16B loads/lane.
__device__ __forceinline__ v16h load_a_row(const _Float16* __restrict__ arow, int kbase) {
  union { v16h v; v8h h[2]; } u;
  u.h[0] = *(const v8h*)(arow + kbase);
  u.h[1] = *(const v8h*)(arow + kbase + 16);
  return u.v;
}

// B-matrix (32x16 f16): lane L -> N=L%16, half=L/16; element e: K = (e<8 ? e : e+8) + 8*half.
// Weights pre-packed so each lane reads one contiguous 32B v16h.
__global__ void pack_b(const float* __restrict__ W, _Float16* __restrict__ out,
                       int N, int K) {
  int idx = blockIdx.x * blockDim.x + threadIdx.x;
  if (idx >= N * K) return;
  int n = idx / K, k = idx % K;
  int nt = n >> 4, kt = k >> 5;
  int kk = k & 31;
  int half = (kk >> 3) & 1, e = (kk & 7) + ((kk >> 4) << 3);
  int lane = (n & 15) + 16 * half;
  int NT = N >> 4;
  size_t off = ((size_t)(kt * NT + nt) * 32 + lane) * 16 + e;
  out[off] = (_Float16)W[(size_t)n * K + k];
}

__global__ void vadd(const float* __restrict__ a, const float* __restrict__ b,
                     float* __restrict__ o, int n) {
  int i = blockIdx.x * blockDim.x + threadIdx.x;
  if (i < n) o[i] = a[i] + b[i];
}

// ---------------- fused ConvSummary (stride==kernel -> per-segment GEMMs) ----
__global__ __launch_bounds__(512)
void summarize_k(const float* __restrict__ xr, const float* __restrict__ Wc1,
                 const float* __restrict__ bc1, const float* __restrict__ Wc2,
                 const float* __restrict__ bc2, _Float16* __restrict__ xo, int shift) {
  __shared__ float xs[SEG];
  __shared__ float h1s[16 * KK];
  int b = blockIdx.x / TT, t = blockIdx.x % TT;
  const float* seg = xr + (size_t)b * LL + (size_t)t * SEG;
  for (int i = threadIdx.x; i < SEG; i += 512) xs[i] = seg[i];
  __syncthreads();
  if (threadIdx.x < 16 * KK) {
    int ci = threadIdx.x / KK, p = threadIdx.x % KK;
    float a = bc1[ci];
    for (int k = 0; k < KK; k++) a += xs[p * KK + k] * Wc1[ci * KK + k];
    h1s[ci * KK + p] = a;
  }
  __syncthreads();
  if (threadIdx.x < CSS) {
    int c = threadIdx.x;
    float a = bc2[c];
    for (int ci = 0; ci < 16; ci++)
      for (int k = 0; k < KK; k++)
        a += h1s[ci * KK + k] * Wc2[(c * 16 + ci) * KK + k];
    if (!shift) {
      xo[((size_t)b * TT + t) * CSS + c] = (_Float16)a;
    } else {  // teacher forcing: dec_in[t+1] = s[t], dec_in[0] = 0
      if (t + 1 < TT) xo[((size_t)b * TT + t + 1) * CSS + c] = (_Float16)a;
      if (t == 0)     xo[((size_t)b * TT) * CSS + c] = (_Float16)0.f;
    }
  }
}

// ---------------- grid-wide barrier for the persistent LSTM --------------
__device__ __forceinline__ void gridbar(unsigned* cnt, unsigned* phase) {
  __syncthreads();
  if (threadIdx.x == 0) {
    __threadfence();                 // release stores
    atomicAdd(cnt, 1u);
    *phase += 1;
    unsigned tgt = (*phase) * (unsigned)NGROUPS;
    while (__hip_atomic_load(cnt, __ATOMIC_ACQUIRE, __HIP_MEMORY_SCOPE_AGENT) < tgt)
      __builtin_amdgcn_s_sleep(1);
  }
  __syncthreads();
}

__device__ __forceinline__ float sigmoidf_(float x) { return 1.f / (1.f + __expf(-x)); }

// Software-pipelined 1-deep: tiles for kt+1 are in flight while the WMMAs for
// kt issue, so the s_wait before each WMMA covers loads that already aged one
// iteration.  unroll_count(2) doubles WMMA issue density in steady state.
__device__ __forceinline__ void mm_acc2048(const _Float16* __restrict__ Abase, int astride,
                                           int mrow, int half, int kts,
                                           const _Float16* __restrict__ Bp, int nt0,
                                           int lane, v8f& acc0, v8f& acc1) {
  const _Float16* arow = Abase + (size_t)mrow * astride + 8 * half;
  v16h a  = load_a_row(arow, 0);
  v16h b0 = *(const v16h*)(Bp + ((size_t)nt0 * 32 + lane) * 16);
  v16h b1 = *(const v16h*)(Bp + ((size_t)(nt0 + 1) * 32 + lane) * 16);
#pragma clang loop unroll_count(2)
  for (int kt = 0; kt < kts; kt++) {
    int ktn = (kt + 1 < kts) ? kt + 1 : 0;
    v16h an  = load_a_row(arow, ktn * 32);
    v16h b0n = *(const v16h*)(Bp + ((size_t)(ktn * NT2048 + nt0    ) * 32 + lane) * 16);
    v16h b1n = *(const v16h*)(Bp + ((size_t)(ktn * NT2048 + nt0 + 1) * 32 + lane) * 16);
    acc0 = __builtin_amdgcn_wmma_f32_16x16x32_f16(false, a, false, b0, (short)0, acc0, false, false);
    acc1 = __builtin_amdgcn_wmma_f32_16x16x32_f16(false, a, false, b1, (short)0, acc1, false, false);
    a = an; b0 = b0n; b1 = b1n;
  }
}

// ---------------- persistent 2-layer LSTM -------------------------------
// 16 workgroups x 512 threads (16 waves).  Workgroup wg owns hidden cols
// [wg*32, wg*32+32).  Wave w: gate = w>>2 (i,f,g,o), mt = w&3 (batch rows
// mt*16..+15).  Cell state c lives in gate-0 waves' VGPRs across all T steps.
__global__ __launch_bounds__(512, 1)
void lstm2_persist(const _Float16* __restrict__ x,       // [B,T,CS] f16
                   const _Float16* __restrict__ pWih0, const _Float16* __restrict__ pWhh0,
                   const _Float16* __restrict__ pWih1, const _Float16* __restrict__ pWhh1,
                   const float* __restrict__ bsum0, const float* __restrict__ bsum1,
                   const float* __restrict__ h_init, const float* __restrict__ c_init, // [2][B,H]
                   float* __restrict__ h_out, float* __restrict__ c_out,               // [2][B,H]
                   _Float16* __restrict__ hws,  // [5][B*H] f16: l0x2, l1x2, h0stage
                   _Float16* __restrict__ y,    // [B,T,H] f16 rows b*T+t, or null
                   unsigned* __restrict__ barcnt) {
  __shared__ float glds[4][64][33];
  const int tid  = threadIdx.x;
  const int lane = tid & 31;
  const int w    = tid >> 5;
  const int gate = w >> 2;
  const int mt   = w & 3;
  const int wg   = blockIdx.x;
  const int j0   = wg * 32;
  const int half = lane >> 4;
  const int lm   = lane & 15;
  unsigned phase = 0;

  // init h buffers (parity 0) from f32 initial state
  for (int i = blockIdx.x * blockDim.x + tid; i < 2 * BB * HH; i += NGROUPS * 512) {
    int l = i / (BB * HH), r = i - l * (BB * HH);
    hws[(size_t)(l * 2) * BH + r] = (_Float16)h_init[i];
  }
  // c state into registers (gate-0 waves own it)
  v8f cst[2][2];
  if (gate == 0) {
    for (int l = 0; l < 2; l++)
      for (int n = 0; n < 2; n++)
        for (int r = 0; r < 8; r++) {
          int Mr = mt * 16 + r + 8 * half;
          int col = j0 + 16 * n + lm;
          cst[l][n][r] = c_init[(size_t)l * BH + (size_t)Mr * HH + col];
        }
  }
  gridbar(barcnt, &phase);

  const int nt0 = gate * 32 + wg * 2;
  const int mrow = mt * 16 + lm;
#pragma clang loop unroll(disable)
  for (int t = 0; t < TT; t++) {
    const int par = t & 1;
    const _Float16* h0prev = hws + (size_t)par * BH;
    _Float16*       h0next = hws + (size_t)(par ^ 1) * BH;
    const _Float16* h1prev = hws + (size_t)(2 + par) * BH;
    _Float16*       h1next = hws + (size_t)(2 + (par ^ 1)) * BH;
    _Float16*       h0st   = hws + (size_t)4 * BH;
    // opaque zero offset per step: weight-tile loads stay inside the t-loop
    // (no LICM hoist + spill) but remain provably global.
    size_t oz = opaque_zero();
    const _Float16* wih0 = pWih0 + oz;
    const _Float16* whh0 = pWhh0 + oz;
    const _Float16* wih1 = pWih1 + oz;
    const _Float16* whh1 = pWhh1 + oz;

    // ================= layer 0 =================
    {
      v8f acc0 = {0,0,0,0,0,0,0,0};
      v8f acc1 = {0,0,0,0,0,0,0,0};
      mm_acc2048(x + (size_t)t * CSS, TT * CSS, mrow, half, 2,  wih0, nt0, lane, acc0, acc1);
      mm_acc2048(h0prev,              HH,       mrow, half, 16, whh0, nt0, lane, acc0, acc1);
      float bb0 = bsum0[gate * HH + j0 + lm];
      float bb1 = bsum0[gate * HH + j0 + 16 + lm];
      for (int r = 0; r < 8; r++) {
        glds[gate][mt * 16 + r + 8 * half][lm]      = acc0[r] + bb0;
        glds[gate][mt * 16 + r + 8 * half][16 + lm] = acc1[r] + bb1;
      }
    }
    __syncthreads();
    if (gate == 0) {
      for (int n = 0; n < 2; n++)
        for (int r = 0; r < 8; r++) {
          int Mr = mt * 16 + r + 8 * half;
          int cl = 16 * n + lm;
          float ig = sigmoidf_(glds[0][Mr][cl]);
          float fg = sigmoidf_(glds[1][Mr][cl]);
          float gg = tanhf(glds[2][Mr][cl]);
          float og = sigmoidf_(glds[3][Mr][cl]);
          float cv = fg * cst[0][n][r] + ig * gg;
          cst[0][n][r] = cv;
          _Float16 hv = (_Float16)(og * tanhf(cv));
          int col = j0 + cl;
          h0next[(size_t)Mr * HH + col] = hv;
          h0st[(size_t)Mr * HH + col]   = hv;
        }
    }
    __syncthreads();
    gridbar(barcnt, &phase);

    // ================= layer 1 =================
    {
      v8f acc0 = {0,0,0,0,0,0,0,0};
      v8f acc1 = {0,0,0,0,0,0,0,0};
      mm_acc2048(h0st,   HH, mrow, half, 16, wih1, nt0, lane, acc0, acc1);
      mm_acc2048(h1prev, HH, mrow, half, 16, whh1, nt0, lane, acc0, acc1);
      float bb0 = bsum1[gate * HH + j0 + lm];
      float bb1 = bsum1[gate * HH + j0 + 16 + lm];
      for (int r = 0; r < 8; r++) {
        glds[gate][mt * 16 + r + 8 * half][lm]      = acc0[r] + bb0;
        glds[gate][mt * 16 + r + 8 * half][16 + lm] = acc1[r] + bb1;
      }
    }
    __syncthreads();
    if (gate == 0) {
      for (int n = 0; n < 2; n++)
        for (int r = 0; r < 8; r++) {
          int Mr = mt * 16 + r + 8 * half;
          int cl = 16 * n + lm;
          float ig = sigmoidf_(glds[0][Mr][cl]);
          float fg = sigmoidf_(glds[1][Mr][cl]);
          float gg = tanhf(glds[2][Mr][cl]);
          float og = sigmoidf_(glds[3][Mr][cl]);
          float cv = fg * cst[1][n][r] + ig * gg;
          cst[1][n][r] = cv;
          _Float16 hv = (_Float16)(og * tanhf(cv));
          int col = j0 + cl;
          h1next[(size_t)Mr * HH + col] = hv;
          if (y) y[((size_t)Mr * TT + t) * HH + col] = hv;
        }
    }
    __syncthreads();
    gridbar(barcnt, &phase);
  }

  // final states: T even -> final h is in parity-0 buffers
  for (int l = 0; l < 2; l++)
    for (int i = blockIdx.x * blockDim.x + tid; i < BB * HH; i += NGROUPS * 512)
      h_out[(size_t)l * BH + i] = (float)hws[(size_t)(l * 2) * BH + i];
  if (gate == 0) {
    for (int l = 0; l < 2; l++)
      for (int n = 0; n < 2; n++)
        for (int r = 0; r < 8; r++) {
          int Mr = mt * 16 + r + 8 * half;
          int col = j0 + 16 * n + lm;
          c_out[(size_t)l * BH + (size_t)Mr * HH + col] = cst[l][n][r];
        }
  }
}

// ---------------- WMMA GEMM: C[M,N] = A_f16[M,K] * Bpacked + bias ----------
__global__ __launch_bounds__(256)
void gemm_a16_bpk(const _Float16* __restrict__ A, const _Float16* __restrict__ Bp,
                  const float* __restrict__ bias, float* __restrict__ C,
                  int M, int N, int K) {
  int gw = (blockIdx.x * blockDim.x + threadIdx.x) >> 5;
  int lane = threadIdx.x & 31;
  int Ntiles = N >> 4;
  int mt = gw / Ntiles, nt = gw % Ntiles;
  if (mt * 16 >= M) return;
  int half = lane >> 4, lm = lane & 15;
  const _Float16* arow = A + (size_t)(mt * 16 + lm) * K + 8 * half;
  v8f acc = {0,0,0,0,0,0,0,0};
  int kts = K >> 5;
  v16h a = load_a_row(arow, 0);
  v16h b = *(const v16h*)(Bp + ((size_t)nt * 32 + lane) * 16);
#pragma clang loop unroll_count(2)
  for (int kt = 0; kt < kts; kt++) {
    int ktn = (kt + 1 < kts) ? kt + 1 : 0;
    v16h an = load_a_row(arow, ktn * 32);
    v16h bn = *(const v16h*)(Bp + ((size_t)(ktn * Ntiles + nt) * 32 + lane) * 16);
    acc = __builtin_amdgcn_wmma_f32_16x16x32_f16(false, a, false, b, (short)0, acc, false, false);
    a = an; b = bn;
  }
  int col = nt * 16 + lm;
  for (int r = 0; r < 8; r++) {
    int row = mt * 16 + r + 8 * half;
    C[(size_t)row * N + col] = acc[r] + bias[col];
  }
}

// ---------------- fused desummarize (two deconvs per step) -----------------
__global__ __launch_bounds__(512)
void desum_k(const float* __restrict__ lat, const float* __restrict__ Wd1,
             const float* __restrict__ bd1, const float* __restrict__ Wd2,
             const float* __restrict__ bd2, float* __restrict__ out) {
  __shared__ float ls[CSS];
  __shared__ float d1s[16 * KK];
  int b = blockIdx.x / TT, t = blockIdx.x % TT;
  if (threadIdx.x < CSS) ls[threadIdx.x] = lat[((size_t)b * TT + t) * CSS + threadIdx.x];
  __syncthreads();
  if (threadIdx.x < 16 * KK) {
    int o = threadIdx.x / KK, p = threadIdx.x % KK;
    float a = bd1[o];
    for (int c = 0; c < CSS; c++) a += ls[c] * Wd1[(c * 16 + o) * KK + p];
    d1s[o * KK + p] = a;
  }
  __syncthreads();
  if (threadIdx.x < SEG) {
    int p = threadIdx.x / KK, j = threadIdx.x % KK;
    float a = bd2[0];
    for (int o = 0; o < 16; o++) a += d1s[o * KK + p] * Wd2[o * KK + j];
    out[(size_t)b * LL + (size_t)t * SEG + threadIdx.x] = a;
  }
}

// ---------------- host driver ----------------
extern "C" void kernel_launch(void* const* d_in, const int* in_sizes, int n_in,
                              void* d_out, int out_size, void* d_ws, size_t ws_size,
                              hipStream_t stream) {
  (void)in_sizes; (void)n_in; (void)out_size; (void)ws_size;
  const float* prev = (const float*)d_in[0];
  const float* next = (const float*)d_in[1];
  const float* Wc1  = (const float*)d_in[2];  const float* bc1 = (const float*)d_in[3];
  const float* Wc2  = (const float*)d_in[4];  const float* bc2 = (const float*)d_in[5];
  const float* Wd1  = (const float*)d_in[6];  const float* bd1 = (const float*)d_in[7];
  const float* Wd2  = (const float*)d_in[8];  const float* bd2 = (const float*)d_in[9];
  const float* eWih0 = (const float*)d_in[10]; const float* eWhh0 = (const float*)d_in[11];
  const float* eBih0 = (const float*)d_in[12]; const float* eBhh0 = (const float*)d_in[13];
  const float* eWih1 = (const float*)d_in[14]; const float* eWhh1 = (const float*)d_in[15];
  const float* eBih1 = (const float*)d_in[16]; const float* eBhh1 = (const float*)d_in[17];
  const float* dWih0 = (const float*)d_in[18]; const float* dWhh0 = (const float*)d_in[19];
  const float* dBih0 = (const float*)d_in[20]; const float* dBhh0 = (const float*)d_in[21];
  const float* dWih1 = (const float*)d_in[22]; const float* dWhh1 = (const float*)d_in[23];
  const float* dBih1 = (const float*)d_in[24]; const float* dBhh1 = (const float*)d_in[25];
  const float* Wfc   = (const float*)d_in[26]; const float* bfc   = (const float*)d_in[27];

  char* ws = (char*)d_ws;
  size_t off = 0;
  auto AL = [&](size_t bytes) -> char* {
    char* p = ws + off;
    off = (off + bytes + 255) & ~(size_t)255;
    return p;
  };
  _Float16* xf_enc   = (_Float16*)AL((size_t)BB * TT * CSS * 2);
  _Float16* xf_dec   = (_Float16*)AL((size_t)BB * TT * CSS * 2);
  _Float16* pk_eWih0 = (_Float16*)AL((size_t)G4 * CSS * 2);
  _Float16* pk_eWhh0 = (_Float16*)AL((size_t)G4 * HH * 2);
  _Float16* pk_eWih1 = (_Float16*)AL((size_t)G4 * HH * 2);
  _Float16* pk_eWhh1 = (_Float16*)AL((size_t)G4 * HH * 2);
  _Float16* pk_dWih0 = (_Float16*)AL((size_t)G4 * CSS * 2);
  _Float16* pk_dWhh0 = (_Float16*)AL((size_t)G4 * HH * 2);
  _Float16* pk_dWih1 = (_Float16*)AL((size_t)G4 * HH * 2);
  _Float16* pk_dWhh1 = (_Float16*)AL((size_t)G4 * HH * 2);
  _Float16* pk_Wfc   = (_Float16*)AL((size_t)CSS * HH * 2);
  float* bsum_e0 = (float*)AL(G4 * 4);
  float* bsum_e1 = (float*)AL(G4 * 4);
  float* bsum_d0 = (float*)AL(G4 * 4);
  float* bsum_d1 = (float*)AL(G4 * 4);
  float* init_h  = (float*)AL((size_t)2 * BH * 4);
  float* init_c  = (float*)AL((size_t)2 * BH * 4);
  float* st_e_h  = (float*)AL((size_t)2 * BH * 4);
  float* st_e_c  = (float*)AL((size_t)2 * BH * 4);
  float* st_d_h  = (float*)AL((size_t)2 * BH * 4);
  float* st_d_c  = (float*)AL((size_t)2 * BH * 4);
  _Float16* hws  = (_Float16*)AL((size_t)5 * BH * 2);
  _Float16* y_dec = (_Float16*)AL((size_t)BB * TT * HH * 2);
  float* lat     = (float*)AL((size_t)BB * TT * CSS * 4);
  unsigned* ctr  = (unsigned*)AL(256);
  unsigned* ctr0 = ctr;
  unsigned* ctr1 = (unsigned*)((char*)ctr + 64);

  // deterministic init (ws is poisoned 0xAA by harness)
  hipMemsetAsync(ctr, 0, 256, stream);
  hipMemsetAsync(init_h, 0, (size_t)2 * BH * 4, stream);
  hipMemsetAsync(init_c, 0, (size_t)2 * BH * 4, stream);

  // fold LSTM biases (bih + bhh)
  vadd<<<(G4 + 255) / 256, 256, 0, stream>>>(eBih0, eBhh0, bsum_e0, G4);
  vadd<<<(G4 + 255) / 256, 256, 0, stream>>>(eBih1, eBhh1, bsum_e1, G4);
  vadd<<<(G4 + 255) / 256, 256, 0, stream>>>(dBih0, dBhh0, bsum_d0, G4);
  vadd<<<(G4 + 255) / 256, 256, 0, stream>>>(dBih1, dBhh1, bsum_d1, G4);

  // pack weights (W[N,K] -> WMMA B tiles of W^T, f16)
  pack_b<<<(G4 * CSS + 255) / 256, 256, 0, stream>>>(eWih0, pk_eWih0, G4, CSS);
  pack_b<<<(G4 * HH  + 255) / 256, 256, 0, stream>>>(eWhh0, pk_eWhh0, G4, HH);
  pack_b<<<(G4 * HH  + 255) / 256, 256, 0, stream>>>(eWih1, pk_eWih1, G4, HH);
  pack_b<<<(G4 * HH  + 255) / 256, 256, 0, stream>>>(eWhh1, pk_eWhh1, G4, HH);
  pack_b<<<(G4 * CSS + 255) / 256, 256, 0, stream>>>(dWih0, pk_dWih0, G4, CSS);
  pack_b<<<(G4 * HH  + 255) / 256, 256, 0, stream>>>(dWhh0, pk_dWhh0, G4, HH);
  pack_b<<<(G4 * HH  + 255) / 256, 256, 0, stream>>>(dWih1, pk_dWih1, G4, HH);
  pack_b<<<(G4 * HH  + 255) / 256, 256, 0, stream>>>(dWhh1, pk_dWhh1, G4, HH);
  pack_b<<<(CSS * HH + 255) / 256, 256, 0, stream>>>(Wfc, pk_Wfc, CSS, HH);

  // summarize prev -> encoder input; next (shifted) -> decoder input
  summarize_k<<<BB * TT, 512, 0, stream>>>(prev, Wc1, bc1, Wc2, bc2, xf_enc, 0);
  summarize_k<<<BB * TT, 512, 0, stream>>>(next, Wc1, bc1, Wc2, bc2, xf_dec, 1);

  // encoder: zero init, states out; decoder: encoder states in, y out
  lstm2_persist<<<NGROUPS, 512, 0, stream>>>(xf_enc, pk_eWih0, pk_eWhh0, pk_eWih1, pk_eWhh1,
                                             bsum_e0, bsum_e1, init_h, init_c,
                                             st_e_h, st_e_c, hws, nullptr, ctr0);
  lstm2_persist<<<NGROUPS, 512, 0, stream>>>(xf_dec, pk_dWih0, pk_dWhh0, pk_dWih1, pk_dWhh1,
                                             bsum_d0, bsum_d1, st_e_h, st_e_c,
                                             st_d_h, st_d_c, hws, y_dec, ctr1);

  // fc: [B*T,512] x [512,64] + bfc
  {
    int M = BB * TT, N = CSS;
    int waves = (M / 16) * (N / 16);
    int blocks = (waves * 32 + 255) / 256;
    gemm_a16_bpk<<<blocks, 256, 0, stream>>>(y_dec, pk_Wfc, bfc, lat, M, N, HH);
  }

  // desummarize -> output waveform
  desum_k<<<BB * TT, 512, 0, stream>>>(lat, Wd1, bd1, Wd2, bd2, (float*)d_out);
}
